// DYConv2d_72318659330761
// MI455X (gfx1250) — matrix operand order repaired and checked by reading, not compiled, not run on hardware
//
#include <hip/hip_runtime.h>

typedef __attribute__((ext_vector_type(16))) __bf16 v16bf;
typedef __attribute__((ext_vector_type(8)))  __bf16 v8bf;
typedef __attribute__((ext_vector_type(8)))  float  v8f;

#define BATCH 32
#define CIN   256
#define COUT  256
#define HW    3136
#define WDIM  56
#define PADW  58
#define PADHW (PADW * PADW)   // 3364

#if __has_builtin(__builtin_amdgcn_global_load_async_to_lds_b128)
#define USE_ASYNC 1
#else
#define USE_ASYNC 0
#endif

#if USE_ASYNC
#if __has_builtin(__builtin_amdgcn_s_wait_asynccnt)
#define WAIT_ASYNC0() __builtin_amdgcn_s_wait_asynccnt(0)
#else
#define WAIT_ASYNC0() asm volatile("s_wait_asynccnt 0x0" ::: "memory")
#endif
// clang diagnostic says param type is: int __attribute__((vector_size(16))) * (device AS)
typedef int i32x4_g __attribute__((vector_size(16)));
#define ASYNC_LDS_B128(g, l, off) \
  __builtin_amdgcn_global_load_async_to_lds_b128( \
      (__attribute__((address_space(1))) i32x4_g*)(g), \
      (__attribute__((address_space(3))) i32x4_g*)(l), (off), 0)
#endif

// ---------------------------------------------------------------------------
// Kernel 1: attention prologue. One block per batch sample.
// ---------------------------------------------------------------------------
__global__ __launch_bounds__(256) void dyconv_attn(
    const float* __restrict__ x,
    const float* __restrict__ fc_share_w, const float* __restrict__ fc_share_b,
    const float* __restrict__ fc_inp_w,  const float* __restrict__ fc_inp_b,
    const float* __restrict__ fc_oup_w,  const float* __restrict__ fc_oup_b,
    const float* __restrict__ fc_k_w,    const float* __restrict__ fc_k_b,
    float* __restrict__ g_ainp, float* __restrict__ g_aoup, float* __restrict__ g_ak)
{
  __shared__ float sPool[CIN];
  __shared__ float sH[16];
  const int b    = blockIdx.x;
  const int t    = threadIdx.x;
  const int lane = t & 31;
  const int wave = t >> 5;
  const float* xb = x + (size_t)b * CIN * HW;

  for (int g = 0; g < CIN / 8; ++g) {
    const int c = g * 8 + wave;
    const float* xc = xb + (size_t)c * HW;
    float sum = 0.f;
    for (int p = lane; p < HW; p += 32) sum += xc[p];
    #pragma unroll
    for (int off = 16; off > 0; off >>= 1) sum += __shfl_down(sum, off, 32);
    if (lane == 0) sPool[c] = sum * (1.0f / (float)HW);
  }
  __syncthreads();

  if (t < 16) {
    float acc = fc_share_b[t];
    const float* wr = fc_share_w + t * CIN;
    for (int c = 0; c < CIN; ++c) acc += sPool[c] * wr[c];
    sH[t] = fmaxf(acc, 0.f);
  }
  __syncthreads();

  float h[16];
  #pragma unroll
  for (int r = 0; r < 16; ++r) h[r] = sH[r];

  float ai = fc_inp_b[t];
  float ao = fc_oup_b[t];
  #pragma unroll
  for (int r = 0; r < 16; ++r) {
    ai += h[r] * fc_inp_w[t * 16 + r];
    ao += h[r] * fc_oup_w[t * 16 + r];
  }
  g_ainp[b * CIN + t]  = 1.f / (1.f + __expf(-ai));
  g_aoup[b * COUT + t] = 1.f / (1.f + __expf(-ao));
  if (t < 9) {
    float ak = fc_k_b[t];
    #pragma unroll
    for (int r = 0; r < 16; ++r) ak += h[r] * fc_k_w[t * 16 + r];
    g_ak[b * 9 + t] = 1.f / (1.f + __expf(-ak));
  }
}

// ---------------------------------------------------------------------------
// Kernel 2a: precompute rank-1-scaled bf16 weights: wbf[b][kk][o][c]
// ---------------------------------------------------------------------------
__global__ __launch_bounds__(256) void build_wbf(
    const float* __restrict__ base_w,
    const float* __restrict__ g_ainp, const float* __restrict__ g_aoup,
    const float* __restrict__ g_ak, __bf16* __restrict__ wbf)
{
  const int chunk = blockIdx.x;   // 0..15 (4096 elems each)
  const int kk    = blockIdx.y;   // 0..8
  const int b     = blockIdx.z;   // 0..31
  const int t     = threadIdx.x;
  const float ak  = g_ak[b * 9 + kk];
  const int base  = chunk * 4096 + t * 16;
  const int o     = base >> 8;
  const int c0    = base & 255;
  const float ao  = g_aoup[b * COUT + o] * ak;
  __bf16* dst = wbf + (size_t)(b * 9 + kk) * (COUT * CIN) + base;
  const float* src = base_w + (size_t)(o * CIN + c0) * 9 + kk;
  const float* ain = g_ainp + b * CIN + c0;
  #pragma unroll
  for (int i = 0; i < 16; ++i)
    dst[i] = (__bf16)(src[(size_t)i * 9] * ao * ain[i]);
}

// ---------------------------------------------------------------------------
// Kernel 2b: zero-padded bf16 input: xpad[b][c][58][58]
// ---------------------------------------------------------------------------
__global__ __launch_bounds__(256) void build_xpad(
    const float* __restrict__ x, __bf16* __restrict__ xpad)
{
  const int bc = blockIdx.x;
  const int t  = threadIdx.x;
  const float* src = x + (size_t)bc * HW;
  __bf16* dst = xpad + (size_t)bc * PADHW;
  for (int i = t; i < PADHW; i += 256) dst[i] = (__bf16)0.f;
  __syncthreads();
  for (int i = t; i < HW; i += 256) {
    const int r = i / WDIM;
    const int cc = i - r * WDIM;
    dst[(r + 1) * PADW + (cc + 1)] = (__bf16)src[i];
  }
}

// ---------------------------------------------------------------------------
// Kernel 3 (fast): double-buffered, async-staged implicit-GEMM, bf16 WMMA.
// Block tile 128(M) x 224(N); 72 K-steps (9 taps x 8 channel-chunks of 32).
// 8 waves = 4(M) x 2(N); each wave: 32x112 = 2x7 accums, 14 WMMA / K-step.
// ---------------------------------------------------------------------------
#define M_TILE 128
#define N_TILE 224
#define KSTEP  32
#define LDA    40
#define LDB    40
#define NSTEPS 72

__global__ __launch_bounds__(256) void dyconv_wmma_fast(
    const __bf16* __restrict__ xpad, const __bf16* __restrict__ wbf,
    float* __restrict__ out)
{
  __shared__ __bf16 As[2][M_TILE * LDA];   // 2 x 10240 B
  __shared__ __bf16 Bs[2][N_TILE * LDB];   // 2 x 17920 B

  const int nt = blockIdx.x;            // 0..13
  const int mt = blockIdx.y;            // 0..1
  const int b  = blockIdx.z;            // 0..31
  const int t  = threadIdx.x;
  const int lane  = t & 31;
  const int wid   = t >> 5;
  const int waveM = wid >> 1;           // 0..3 -> 32 rows
  const int waveN = wid & 1;            // 0..1 -> 112 cols
  const int oBase = mt * M_TILE;
  const int pBase = nt * N_TILE;

  const int aRow  = t >> 1;             // 0..127
  const int aHalf = (t & 1) * 16;       // 0 or 16
  const int aLds  = aRow * LDA + aHalf;

  const __bf16* wb = wbf + (size_t)b * 9 * (COUT * CIN);
  const __bf16* xb = xpad + (size_t)b * CIN * PADHW;

  // B staging coordinates, packed: [31:14] = c*PADHW + padBase, [13:0] = col*LDB + c
  unsigned int bPk[28];
  #pragma unroll
  for (int i = 0; i < 28; ++i) {
    const int idx = t + (i << 8);           // 32(c) x 224(col)
    const int c   = idx / N_TILE;
    const int col = idx - c * N_TILE;
    const int ry  = col / WDIM;
    const int rx  = col - ry * WDIM;
    const unsigned int xoff = c * PADHW + (nt * 4 + ry + 1) * PADW + (rx + 1);
    const unsigned int loff = col * LDB + c;
    bPk[i] = (xoff << 14) | loff;
  }

  const v8f zero8 = {0.f, 0.f, 0.f, 0.f, 0.f, 0.f, 0.f, 0.f};
  v8f acc[2][7];
  #pragma unroll
  for (int mi = 0; mi < 2; ++mi)
    #pragma unroll
    for (int ni = 0; ni < 7; ++ni) acc[mi][ni] = zero8;

  const int kLoA     = (lane < 16) ? 0 : 8;
  const int kOffB    = (lane < 16) ? 0 : 8;
  const int halfLane = lane & 15;

  // ---- prologue: stage step 0 into buffer 0 ----
  {
    const __bf16* srcA = wb + (size_t)(oBase + aRow) * CIN + aHalf;  // kk=0, kc=0
#if USE_ASYNC
    ASYNC_LDS_B128(srcA, &As[0][aLds], 0);
    ASYNC_LDS_B128(srcA, &As[0][aLds], 16);
#else
    const uint4 q0 = ((const uint4*)srcA)[0];
    const uint4 q1 = ((const uint4*)srcA)[1];
    uint4* dstA = (uint4*)&As[0][aLds];
    dstA[0] = q0; dstA[1] = q1;
#endif
    const __bf16* xs = xb + (-PADW - 1);    // kk=0 -> dy=-1,dx=-1; kc=0
    #pragma unroll
    for (int i = 0; i < 28; ++i) {
      const unsigned int pk = bPk[i];
      Bs[0][pk & 16383u] = xs[pk >> 14];
    }
  }

  for (int s = 0; s < NSTEPS; ++s) {
    const int cur = s & 1;
#if USE_ASYNC
    WAIT_ASYNC0();
#endif
    __syncthreads();

    const int haveNext = (s + 1 < NSTEPS);
    unsigned short bReg[28];
#if !USE_ASYNC
    uint4 aReg0, aReg1;
#endif
    if (haveNext) {
      const int s1  = s + 1;
      const int kk1 = s1 >> 3;
      const int kc1 = (s1 & 7) << 5;
      const int dy1 = kk1 / 3 - 1;
      const int dx1 = kk1 % 3 - 1;
      const __bf16* srcA = wb + (size_t)kk1 * (COUT * CIN)
                              + (size_t)(oBase + aRow) * CIN + kc1 + aHalf;
#if USE_ASYNC
      ASYNC_LDS_B128(srcA, &As[cur ^ 1][aLds], 0);
      ASYNC_LDS_B128(srcA, &As[cur ^ 1][aLds], 16);
#else
      __builtin_prefetch(srcA, 0, 0);
      aReg0 = ((const uint4*)srcA)[0];
      aReg1 = ((const uint4*)srcA)[1];
#endif
      const __bf16* xs = xb + (dy1 * PADW + dx1) + (size_t)kc1 * PADHW;
      #pragma unroll
      for (int i = 0; i < 28; ++i)
        bReg[i] = *(const unsigned short*)&xs[bPk[i] >> 14];
    }

    // ---- compute on buf[cur]: fragments + 14 WMMAs ----
    {
      const __bf16* asCur = As[cur];
      const __bf16* bsCur = Bs[cur];
      v16bf aF[2], bF[7];
      #pragma unroll
      for (int mi = 0; mi < 2; ++mi) {
        const int row = waveM * 32 + mi * 16 + halfLane;
        const v8bf lo = *(const v8bf*)&asCur[row * LDA + kLoA];
        const v8bf hi = *(const v8bf*)&asCur[row * LDA + kLoA + 16];
        aF[mi] = __builtin_shufflevector(lo, hi, 0,1,2,3,4,5,6,7,8,9,10,11,12,13,14,15);
      }
      #pragma unroll
      for (int ni = 0; ni < 7; ++ni) {
        const int col = waveN * 112 + ni * 16 + halfLane;
        const v8bf lo = *(const v8bf*)&bsCur[col * LDB + kOffB];
        const v8bf hi = *(const v8bf*)&bsCur[col * LDB + kOffB + 16];
        bF[ni] = __builtin_shufflevector(lo, hi, 0,1,2,3,4,5,6,7,8,9,10,11,12,13,14,15);
      }
      #pragma unroll
      for (int mi = 0; mi < 2; ++mi)
        #pragma unroll
        for (int ni = 0; ni < 7; ++ni)
          acc[mi][ni] = __builtin_amdgcn_wmma_f32_16x16x32_bf16(
              false, aF[mi], false, bF[ni], (short)0, acc[mi][ni], false, false);
    }

    // ---- commit s+1 staging to LDS (after cur's reads) ----
    if (haveNext) {
      __bf16* bsNxt = Bs[cur ^ 1];
      #pragma unroll
      for (int i = 0; i < 28; ++i)
        *(unsigned short*)&bsNxt[bPk[i] & 16383u] = bReg[i];
#if !USE_ASYNC
      uint4* dstA = (uint4*)&As[cur ^ 1][aLds];
      dstA[0] = aReg0; dstA[1] = aReg1;
#endif
    }
  }

  // ---- epilogue ----
  const int mhalf = (lane >> 4) * 8;
  #pragma unroll
  for (int mi = 0; mi < 2; ++mi) {
    #pragma unroll
    for (int ni = 0; ni < 7; ++ni) {
      const int colg = pBase + waveN * 112 + ni * 16 + halfLane;
      #pragma unroll
      for (int j = 0; j < 8; ++j) {
        const int og = oBase + waveM * 32 + mi * 16 + mhalf + j;
        out[((size_t)b * COUT + og) * (size_t)HW + colg] = acc[mi][ni][j];
      }
    }
  }
}

// ---------------------------------------------------------------------------
// Fallback kernel (tiny workspace): fused rescale + GEMM, 128x64 tile.
// ---------------------------------------------------------------------------
__global__ __launch_bounds__(256) void dyconv_wmma_small(
    const float* __restrict__ x, const float* __restrict__ base_w,
    const float* __restrict__ g_ainp, const float* __restrict__ g_aoup,
    const float* __restrict__ g_ak, float* __restrict__ out)
{
  __shared__ __bf16 As[128 * LDA];
  __shared__ __bf16 Bs[64 * LDB];
  __shared__ float  sAinp[CIN];
  __shared__ float  sAoup[128];
  __shared__ float  sAk[9];

  const int nt = blockIdx.x;
  const int mt = blockIdx.y;
  const int b  = blockIdx.z;
  const int t  = threadIdx.x;
  const int lane  = t & 31;
  const int wid   = t >> 5;
  const int waveM = wid & 3;
  const int waveN = wid >> 2;
  const int oBase = mt * 128;
  const int pBase = nt * 64;

  sAinp[t] = g_ainp[b * CIN + t];
  if (t < 128) sAoup[t] = g_aoup[b * COUT + oBase + t];
  if (t < 9)   sAk[t]   = g_ak[b * 9 + t];

  const v8f zero8 = {0.f, 0.f, 0.f, 0.f, 0.f, 0.f, 0.f, 0.f};
  v8f acc[2][2] = {{zero8, zero8}, {zero8, zero8}};

  const int kLoA     = (lane < 16) ? 0 : 8;
  const int kOffB    = (lane < 16) ? 0 : 8;
  const int halfLane = lane & 15;

  for (int kk = 0; kk < 9; ++kk) {
    const int dy = kk / 3 - 1;
    const int dx = kk % 3 - 1;
    for (int kc = 0; kc < CIN; kc += 32) {
      __syncthreads();
      const float akv = sAk[kk];
      #pragma unroll
      for (int i = 0; i < 16; ++i) {
        const int idx = t * 16 + i;
        const int row = idx >> 5, c = idx & 31;
        const float w = base_w[(size_t)(((oBase + row) * CIN + (kc + c)) * 9 + kk)];
        As[row * LDA + c] = (__bf16)(w * akv * sAoup[row] * sAinp[kc + c]);
      }
      #pragma unroll
      for (int i = 0; i < 8; ++i) {
        const int idx = t * 8 + i;
        const int c = idx >> 6, col = idx & 63;
        const int p = pBase + col;
        const int oy = p / WDIM, ox = p - (p / WDIM) * WDIM;
        const int iy = oy + dy, ix = ox + dx;
        float v = 0.f;
        if (iy >= 0 && iy < WDIM && ix >= 0 && ix < WDIM)
          v = x[((size_t)b * CIN + (kc + c)) * HW + iy * WDIM + ix];
        Bs[col * LDB + c] = (__bf16)v;
      }
      __syncthreads();

      v16bf aF[2], bF[2];
      #pragma unroll
      for (int mi = 0; mi < 2; ++mi) {
        const int row = waveM * 32 + mi * 16 + halfLane;
        const v8bf lo = *(const v8bf*)&As[row * LDA + kLoA];
        const v8bf hi = *(const v8bf*)&As[row * LDA + kLoA + 16];
        aF[mi] = __builtin_shufflevector(lo, hi, 0,1,2,3,4,5,6,7,8,9,10,11,12,13,14,15);
      }
      #pragma unroll
      for (int ni = 0; ni < 2; ++ni) {
        const int col = waveN * 32 + ni * 16 + halfLane;
        const v8bf lo = *(const v8bf*)&Bs[col * LDB + kOffB];
        const v8bf hi = *(const v8bf*)&Bs[col * LDB + kOffB + 16];
        bF[ni] = __builtin_shufflevector(lo, hi, 0,1,2,3,4,5,6,7,8,9,10,11,12,13,14,15);
      }
      #pragma unroll
      for (int mi = 0; mi < 2; ++mi)
        #pragma unroll
        for (int ni = 0; ni < 2; ++ni)
          acc[mi][ni] = __builtin_amdgcn_wmma_f32_16x16x32_bf16(
              false, aF[mi], false, bF[ni], (short)0, acc[mi][ni], false, false);
    }
  }

  const int mhalf = (lane >> 4) * 8;
  #pragma unroll
  for (int mi = 0; mi < 2; ++mi)
    #pragma unroll
    for (int ni = 0; ni < 2; ++ni) {
      const int colg = pBase + waveN * 32 + ni * 16 + halfLane;
      #pragma unroll
      for (int j = 0; j < 8; ++j) {
        const int og = oBase + waveM * 32 + mi * 16 + mhalf + j;
        out[((size_t)b * COUT + og) * (size_t)HW + colg] = acc[mi][ni][j];
      }
    }
}

// ---------------------------------------------------------------------------
extern "C" void kernel_launch(void* const* d_in, const int* in_sizes, int n_in,
                              void* d_out, int out_size, void* d_ws, size_t ws_size,
                              hipStream_t stream) {
  const float* x          = (const float*)d_in[0];
  const float* base_w     = (const float*)d_in[1];
  const float* fc_share_w = (const float*)d_in[2];
  const float* fc_share_b = (const float*)d_in[3];
  const float* fc_inp_w   = (const float*)d_in[4];
  const float* fc_inp_b   = (const float*)d_in[5];
  const float* fc_oup_w   = (const float*)d_in[6];
  const float* fc_oup_b   = (const float*)d_in[7];
  const float* fc_k_w     = (const float*)d_in[8];
  const float* fc_k_b     = (const float*)d_in[9];
  float* out = (float*)d_out;

  float* ws     = (float*)d_ws;
  float* g_ainp = ws;
  float* g_aoup = ws + BATCH * CIN;
  float* g_ak   = ws + BATCH * (CIN + COUT);

  const size_t attnBytes = 128 * 1024;
  const size_t wbfBytes  = (size_t)BATCH * 9 * COUT * CIN * 2;   // 37.7 MB
  const size_t xpadBytes = (size_t)BATCH * CIN * PADHW * 2;      // 55.1 MB

  dyconv_attn<<<BATCH, 256, 0, stream>>>(
      x, fc_share_w, fc_share_b, fc_inp_w, fc_inp_b,
      fc_oup_w, fc_oup_b, fc_k_w, fc_k_b, g_ainp, g_aoup, g_ak);

  if (ws_size >= attnBytes + wbfBytes + xpadBytes) {
    __bf16* wbf  = (__bf16*)((char*)d_ws + attnBytes);
    __bf16* xpad = (__bf16*)((char*)d_ws + attnBytes + wbfBytes);

    build_wbf<<<dim3(16, 9, BATCH), 256, 0, stream>>>(base_w, g_ainp, g_aoup, g_ak, wbf);
    build_xpad<<<BATCH * CIN, 256, 0, stream>>>(x, xpad);

    dim3 grid(HW / N_TILE, COUT / M_TILE, BATCH);   // 14 x 2 x 32
    dyconv_wmma_fast<<<grid, 256, 0, stream>>>(xpad, wbf, out);
  } else {
    dim3 grid(HW / 64, COUT / 128, BATCH);          // 49 x 2 x 32
    dyconv_wmma_small<<<grid, 256, 0, stream>>>(x, base_w, g_ainp, g_aoup, g_ak, out);
  }
}
